// NonLocalFeatureBlock_49168785605358
// MI455X (gfx1250) — matrix operand order repaired and checked by reading, not compiled, not run on hardware
//
#include <hip/hip_runtime.h>

typedef __attribute__((ext_vector_type(16))) __bf16 bfrag;
typedef __attribute__((ext_vector_type(8)))  float  v8f;

#define KEPS 1e-3f
#define NPTS 262144
#define KV_COLS 80   // 64 kv columns + 1 ksum column (+15 pad, stays zero)

__device__ __forceinline__ v8f wmma_bf16(bfrag a, bfrag b, v8f c) {
  // D = A(16x32 bf16) * B(32x16 bf16) + C(16x16 f32)
  return __builtin_amdgcn_wmma_f32_16x16x32_bf16(false, a, false, b, (short)0, c,
                                                 false, false);
}

// Build one B-fragment of a 64x64 row-major fp32 weight matrix.
// B layout: col N = lane&15, K = 16*(e>>3) + 8*(lane>>4) + (e&7)  (within the 32-wide k-chunk kk)
__device__ __forceinline__ bfrag make_w_frag(const float* __restrict__ W, int kk,
                                             int j, int half, int l16) {
  bfrag b;
#pragma unroll
  for (int e = 0; e < 16; ++e) {
    int k = kk * 32 + 16 * (e >> 3) + 8 * half + (e & 7);
    b[e] = (__bf16)W[k * 64 + j * 16 + l16];
  }
  return b;
}

// A-fragment of a 16x64 fp32 tile of x; xr = row pointer for this lane's row (m = lane&15).
// A layout: K = 16*(e>>3) + 8*(lane>>4) + (e&7)
__device__ __forceinline__ bfrag load_x_frag(const float* __restrict__ xr, int kk,
                                             int half) {
  bfrag a;
#pragma unroll
  for (int g = 0; g < 2; ++g) {
    int kb = kk * 32 + 16 * g + 8 * half;
    float4 f0 = *(const float4*)(xr + kb);
    float4 f1 = *(const float4*)(xr + kb + 4);
    a[8 * g + 0] = (__bf16)f0.x; a[8 * g + 1] = (__bf16)f0.y;
    a[8 * g + 2] = (__bf16)f0.z; a[8 * g + 3] = (__bf16)f0.w;
    a[8 * g + 4] = (__bf16)f1.x; a[8 * g + 5] = (__bf16)f1.y;
    a[8 * g + 6] = (__bf16)f1.z; a[8 * g + 7] = (__bf16)f1.w;
  }
  return a;
}

__global__ void zero_kv(float* __restrict__ ws) {
  int i = blockIdx.x * 256 + threadIdx.x;
  if (i < 64 * KV_COLS) ws[i] = 0.0f;
}

// Pass 1: kv[m][d] += sum_n kp[n][m]*v[n][d], ksum[m] += sum_n kp[n][m]
__global__ __launch_bounds__(256) void favor_pass1(
    const float* __restrict__ x, const float* __restrict__ Wk,
    const float* __restrict__ bk, const float* __restrict__ Wv,
    const float* __restrict__ bv, float* __restrict__ kv_acc, int rows_per_wg) {
  __shared__ __bf16 kp_s[128][72];
  __shared__ __bf16 v_s[128][72];
  __shared__ bfrag wk_f[2][4][32];  // [kk][j][lane], fragment-ordered
  __shared__ bfrag wv_f[2][4][32];

  const int lane = threadIdx.x & 31;
  const int wave = threadIdx.x >> 5;
  const int half = lane >> 4;
  const int l16 = lane & 15;

  // Cooperative fragment-ordered staging of Wk/Wv (16 fragments over 8 waves).
  for (int idx = wave; idx < 16; idx += 8) {
    const int m = idx >> 3, kk = (idx >> 2) & 1, j = idx & 3;
    bfrag f = make_w_frag(m ? Wv : Wk, kk, j, half, l16);
    bfrag(*dst)[4][32] = m ? wv_f : wk_f;
    dst[kk][j][lane] = f;
  }
  float bkc[4], bvc[4];
#pragma unroll
  for (int j = 0; j < 4; ++j) {
    bkc[j] = bk[j * 16 + l16];
    bvc[j] = bv[j * 16 + l16];
  }
  __syncthreads();

  // Stage-2 tile assignment: each wave owns feat-row ft and 2-3 column tiles.
  const int ft = wave & 3;
  const int dt0 = (wave < 4) ? 0 : 3;
  const int ndt = (wave < 4) ? 3 : 2;
  v8f acc[3];
#pragma unroll
  for (int i = 0; i < 3; ++i) {
    v8f z = {};
    acc[i] = z;
  }

  const int row_base = blockIdx.x * rows_per_wg;
  for (int cs = 0; cs < rows_per_wg; cs += 128) {
    const int row0 = row_base + cs + wave * 16;
    const float* xr = x + (size_t)(row0 + l16) * 64;
    bfrag A0 = load_x_frag(xr, 0, half);
    bfrag A1 = load_x_frag(xr, 1, half);

    // Projections k = x@Wk + bk, v = x@Wv + bv for this wave's 16 rows.
#pragma unroll
    for (int j = 0; j < 4; ++j) {
      v8f ck = {};
      ck = wmma_bf16(A0, wk_f[0][j][lane], ck);
      ck = wmma_bf16(A1, wk_f[1][j][lane], ck);
      v8f cv = {};
      cv = wmma_bf16(A0, wv_f[0][j][lane], cv);
      cv = wmma_bf16(A1, wv_f[1][j][lane], cv);
#pragma unroll
      for (int r = 0; r < 8; ++r) {
        int rl = wave * 16 + r + 8 * half;  // C layout: M = r + 8*half, N = l16
        kp_s[rl][j * 16 + l16] = (__bf16)(fmaxf(ck[r] + bkc[j], 0.0f) + KEPS);
        v_s[rl][j * 16 + l16] = (__bf16)(cv[r] + bvc[j]);
      }
    }
    __syncthreads();

    // kv_ext += kp^T @ [v | ones-col], K = 128 rows in 4 chunks of 32.
#pragma unroll
    for (int c = 0; c < 4; ++c) {
      bfrag a;  // transposed gather: A' = kp^T, shared across this wave's B tiles
#pragma unroll
      for (int e = 0; e < 16; ++e) {
        int k = c * 32 + 16 * (e >> 3) + 8 * half + (e & 7);
        a[e] = kp_s[k][ft * 16 + l16];
      }
#pragma unroll
      for (int ti = 0; ti < 3; ++ti) {
        if (ti < ndt) {
          const int dt = dt0 + ti;
          bfrag b;
          if (dt < 4) {
#pragma unroll
            for (int e = 0; e < 16; ++e) {
              int k = c * 32 + 16 * (e >> 3) + 8 * half + (e & 7);
              b[e] = v_s[k][dt * 16 + l16];
            }
          } else {
            __bf16 one = (__bf16)((l16 == 0) ? 1.0f : 0.0f);  // ksum column
#pragma unroll
            for (int e = 0; e < 16; ++e) b[e] = one;
          }
          acc[ti] = wmma_bf16(a, b, acc[ti]);
        }
      }
    }
    __syncthreads();
  }

  // Flush partials (L2-resident 20KB buffer; ~512 atomics per address total).
#pragma unroll
  for (int ti = 0; ti < 3; ++ti) {
    if (ti < ndt) {
#pragma unroll
      for (int r = 0; r < 8; ++r) {
        atomicAdd(
            &kv_acc[(ft * 16 + r + 8 * half) * KV_COLS + (dt0 + ti) * 16 + l16],
            acc[ti][r]);
      }
    }
  }
}

// Pass 2: out = (qp @ kv) / (qp @ ksum)
__global__ __launch_bounds__(256) void favor_pass2(
    const float* __restrict__ x, const float* __restrict__ Wq,
    const float* __restrict__ bq, const float* __restrict__ kv_acc,
    float* __restrict__ out, int chunks_per_wg) {
  __shared__ __bf16 qp_s[128][72];   // wave-private 16-row regions
  __shared__ bfrag wq_f[2][4][32];   // [kk][j][lane]
  __shared__ bfrag kv_f[2][5][32];   // [c2][dt][lane]

  const int lane = threadIdx.x & 31;
  const int wave = threadIdx.x >> 5;
  const int half = lane >> 4;
  const int l16 = lane & 15;

  // Fragment-ordered staging of Wq (8 frags) and kv_ext (10 frags).
  {
    const int kk = wave >> 2, j = wave & 3;
    wq_f[kk][j][lane] = make_w_frag(Wq, kk, j, half, l16);
  }
  for (int idx = wave; idx < 10; idx += 8) {
    const int c2 = idx / 5, dt = idx % 5;
    bfrag f;
#pragma unroll
    for (int e = 0; e < 16; ++e) {
      int k = c2 * 32 + 16 * (e >> 3) + 8 * half + (e & 7);
      f[e] = (__bf16)kv_acc[k * KV_COLS + dt * 16 + l16];
    }
    kv_f[c2][dt][lane] = f;
  }
  float bqc[4];
#pragma unroll
  for (int j = 0; j < 4; ++j) bqc[j] = bq[j * 16 + l16];
  __syncthreads();

  for (int cs = 0; cs < chunks_per_wg; ++cs) {
    const int row0 = (blockIdx.x * chunks_per_wg + cs) * 128 + wave * 16;
    const float* xr = x + (size_t)(row0 + l16) * 64;
    bfrag A0 = load_x_frag(xr, 0, half);
    bfrag A1 = load_x_frag(xr, 1, half);

    // q projection -> qp; bounce through wave-private LDS (DS is in-order
    // within a wave, and only this wave touches rows [wave*16, wave*16+16)).
#pragma unroll
    for (int j = 0; j < 4; ++j) {
      v8f cq = {};
      cq = wmma_bf16(A0, wq_f[0][j][lane], cq);
      cq = wmma_bf16(A1, wq_f[1][j][lane], cq);
#pragma unroll
      for (int r = 0; r < 8; ++r)
        qp_s[wave * 16 + r + 8 * half][j * 16 + l16] =
            (__bf16)(fmaxf(cq[r] + bqc[j], 0.0f) + KEPS);
    }

    bfrag Q[2];
#pragma unroll
    for (int kk = 0; kk < 2; ++kk)
#pragma unroll
      for (int e = 0; e < 16; ++e) {
        int k = kk * 32 + 16 * (e >> 3) + 8 * half + (e & 7);
        Q[kk][e] = qp_s[wave * 16 + l16][k];
      }

    // [num | denom] = qp @ kv_ext
    v8f C[5];
#pragma unroll
    for (int dt = 0; dt < 5; ++dt) {
      v8f c = {};
      c = wmma_bf16(Q[0], kv_f[0][dt][lane], c);
      c = wmma_bf16(Q[1], kv_f[1][dt][lane], c);
      C[dt] = c;
    }

    // denom is tile 4 column 0 (lanes 0 and 16); broadcast within each half.
    float dn[8];
#pragma unroll
    for (int r = 0; r < 8; ++r) dn[r] = __shfl(C[4][r], lane & 16, 32);

#pragma unroll
    for (int dt = 0; dt < 4; ++dt)
#pragma unroll
      for (int r = 0; r < 8; ++r)
        out[(size_t)(row0 + r + 8 * half) * 64 + dt * 16 + l16] = C[dt][r] / dn[r];
  }
}

extern "C" void kernel_launch(void* const* d_in, const int* in_sizes, int n_in,
                              void* d_out, int out_size, void* d_ws, size_t ws_size,
                              hipStream_t stream) {
  const float* x = (const float*)d_in[0];
  const float* Wq = (const float*)d_in[1];
  const float* bq = (const float*)d_in[2];
  const float* Wk = (const float*)d_in[3];
  const float* bk = (const float*)d_in[4];
  const float* Wv = (const float*)d_in[5];
  const float* bv = (const float*)d_in[6];
  float* out = (float*)d_out;
  float* kv = (float*)d_ws;  // 64*80 fp32 = 20 KB, zeroed each launch

  zero_kv<<<(64 * KV_COLS + 255) / 256, 256, 0, stream>>>(kv);
  const int wgs = 512;  // 512 rows per WG => 4 chunks of 128
  favor_pass1<<<wgs, 256, 0, stream>>>(x, Wk, bk, Wv, bv, kv, NPTS / wgs);
  favor_pass2<<<wgs, 256, 0, stream>>>(x, Wq, bq, kv, out, NPTS / 128 / wgs);
}
